// MultiheadAttentionBlock_14568529068625
// MI455X (gfx1250) — matrix operand order repaired and checked by reading, not compiled, not run on hardware
//
#include <hip/hip_runtime.h>

// ---------------------------------------------------------------------------
// MHA block for MI455X (gfx1250, wave32).
// bf16 WMMA (v_wmma_f32_16x16x32_bf16) for all 4 projections + flash attention.
// fp32 accumulate everywhere; softmax/RoPE in fp32 VALU.
// ---------------------------------------------------------------------------

#define BATCH 2
#define SEQ   2048
#define EMB   2048
#define NH    16
#define HD    128
#define MTOK  (BATCH * SEQ)   // 4096 tokens

typedef __attribute__((ext_vector_type(16))) __bf16 v16bf;
typedef __attribute__((ext_vector_type(8)))  float  v8f;

union Frag {
    v16bf v;
    uint4 q[2];
};

__device__ __forceinline__ v8f wmma_bf16(v16bf a, v16bf b, v8f c) {
    // D = A(16x32) * B(32x16) + C(16x16 f32)
    return __builtin_amdgcn_wmma_f32_16x16x32_bf16(
        false, a, false, b, (short)0, c, false, false);
}

__device__ __forceinline__ float rowmax16(float x) {
    #pragma unroll
    for (int off = 8; off > 0; off >>= 1) x = fmaxf(x, __shfl_xor(x, off, 16));
    return x;
}
__device__ __forceinline__ float rowsum16(float x) {
    #pragma unroll
    for (int off = 8; off > 0; off >>= 1) x += __shfl_xor(x, off, 16);
    return x;
}

// ---------------------------------------------------------------------------
// Elementwise converts
// ---------------------------------------------------------------------------
__global__ __launch_bounds__(256) void cvt_f32_bf16(const float* __restrict__ in,
                                                    __bf16* __restrict__ out, int n) {
    int i = blockIdx.x * 256 + threadIdx.x;
    if (i < n) out[i] = (__bf16)in[i];
}

// W [K x N] row-major fp32  ->  Wt [N x K] row-major bf16 (coalesced writes)
__global__ __launch_bounds__(256) void cvt_w_transpose(const float* __restrict__ W,
                                                       __bf16* __restrict__ Wt,
                                                       int K, int N) {
    int i = blockIdx.x * 256 + threadIdx.x;            // i = n*K + k
    if (i >= K * N) return;
    int k = i % K, n = i / K;
    Wt[(size_t)n * K + k] = (__bf16)W[(size_t)k * N + n];
}

// ---------------------------------------------------------------------------
// bf16 WMMA GEMM:  C[M x N] f32  =  A[M x K] bf16 (row-major) * Bt[N x K] bf16
// Block: 128 threads = 4 waves in 2x2; wave tile 32(M) x 64(N); K step 32.
// Grid: (N/128, M/64)
// ---------------------------------------------------------------------------
__global__ __launch_bounds__(128) void gemm_bf16(const __bf16* __restrict__ A,
                                                 const __bf16* __restrict__ Bt,
                                                 float* __restrict__ C,
                                                 int M, int N, int K) {
    const int wave = threadIdx.x >> 5;
    const int lane = threadIdx.x & 31;
    const int g = lane >> 4;           // half-wave group
    const int r = lane & 15;           // row/col within tile
    const int wm = wave >> 1, wn = wave & 1;
    const int m0 = blockIdx.y * 64 + wm * 32;
    const int n0 = blockIdx.x * 128 + wn * 64;

    v8f acc[2][4];
    #pragma unroll
    for (int i = 0; i < 2; ++i)
        #pragma unroll
        for (int j = 0; j < 4; ++j) acc[i][j] = (v8f)0.0f;

    for (int k0 = 0; k0 < K; k0 += 32) {
        Frag a[2], b[4];
        #pragma unroll
        for (int i = 0; i < 2; ++i) {
            const __bf16* pa = A + (size_t)(m0 + i * 16 + r) * K + k0;
            a[i].q[0] = *(const uint4*)(pa + g * 8);        // K = 8g .. 8g+7
            a[i].q[1] = *(const uint4*)(pa + 16 + g * 8);   // K = 16+8g ..
        }
        #pragma unroll
        for (int j = 0; j < 4; ++j) {
            const __bf16* pb = Bt + (size_t)(n0 + j * 16 + r) * K + k0;
            b[j].q[0] = *(const uint4*)(pb + g * 8);        // K = 8g .. 8g+7
            b[j].q[1] = *(const uint4*)(pb + 16 + g * 8);   // K = 16+8g ..
        }
        #pragma unroll
        for (int i = 0; i < 2; ++i)
            #pragma unroll
            for (int j = 0; j < 4; ++j)
                acc[i][j] = wmma_bf16(a[i].v, b[j].v, acc[i][j]);
    }

    // C layout: lane holds (m = v + 8g, n = r) within each 16x16 tile
    #pragma unroll
    for (int i = 0; i < 2; ++i)
        #pragma unroll
        for (int j = 0; j < 4; ++j)
            #pragma unroll
            for (int v = 0; v < 8; ++v) {
                int m = m0 + i * 16 + v + 8 * g;
                int n = n0 + j * 16 + r;
                C[(size_t)m * N + n] = acc[i][j][v];
            }
}

// ---------------------------------------------------------------------------
// RoPE: fp32 [B,S,E] token-major -> bf16 [B,H,S,D] head-major, rotated halves
// one thread per (b,h,s,i) pair, i in [0,64)
// ---------------------------------------------------------------------------
__global__ __launch_bounds__(256) void rope_to_bf16(const float* __restrict__ X,
                                                    __bf16* __restrict__ O) {
    int tid = blockIdx.x * 256 + threadIdx.x;
    if (tid >= BATCH * NH * SEQ * (HD / 2)) return;
    int i  = tid & 63;
    int s  = (tid >> 6) & (SEQ - 1);
    int bh = tid >> 17;                 // 6 + 11 bits
    int b = bh / NH, h = bh % NH;

    const float* src = X + ((size_t)b * SEQ + s) * EMB + h * HD;
    float x1 = src[i], x2 = src[i + 64];
    // inv_freq = 10000^(-2i/128) = exp(-i * ln(10000)/64)
    float freq = __expf(-(float)i * (9.210340371976184f / 64.0f));
    float ang = (float)s * freq;
    float c = cosf(ang), sn = sinf(ang);
    __bf16* dst = O + ((size_t)bh * SEQ + s) * HD;
    dst[i]      = (__bf16)(x1 * c - x2 * sn);
    dst[i + 64] = (__bf16)(x1 * sn + x2 * c);
}

// V: fp32 [B,S,E] -> bf16 Vt [B,H,D,S]  (contraction dim contiguous for PV WMMA)
__global__ __launch_bounds__(256) void v_transpose(const float* __restrict__ X,
                                                   __bf16* __restrict__ Vt) {
    int tid = blockIdx.x * 256 + threadIdx.x;
    if (tid >= BATCH * NH * HD * SEQ) return;
    int s  = tid & (SEQ - 1);
    int d  = (tid >> 11) & (HD - 1);
    int bh = tid >> 18;
    int b = bh / NH, h = bh % NH;
    Vt[((size_t)bh * HD + d) * SEQ + s] =
        (__bf16)X[((size_t)b * SEQ + s) * EMB + h * HD + d];
}

// ---------------------------------------------------------------------------
// Flash attention. Q,K bf16 [B,H,S,D]; Vt bf16 [B,H,D,S]; out bf16 [B,S,E].
// Block: 128 thr = 4 waves, each wave owns 16 query rows; 32 keys per step.
// Grid: (SEQ/64, BATCH*NH)
// ---------------------------------------------------------------------------
__global__ __launch_bounds__(128) void flash_attn(const __bf16* __restrict__ Q,
                                                  const __bf16* __restrict__ K,
                                                  const __bf16* __restrict__ Vt,
                                                  __bf16* __restrict__ O) {
    const int wave = threadIdx.x >> 5;
    const int lane = threadIdx.x & 31;
    const int g = lane >> 4, r = lane & 15;
    const int bh = blockIdx.y;
    const int b = bh / NH, h = bh % NH;
    const int q0 = blockIdx.x * 64 + wave * 16;

    const __bf16* Qb = Q  + (size_t)bh * SEQ * HD;
    const __bf16* Kb = K  + (size_t)bh * SEQ * HD;
    const __bf16* Vb = Vt + (size_t)bh * HD * SEQ;

    __shared__ __align__(16) __bf16 pstage[4][16 * 32];
    __bf16* pw = pstage[wave];

    // Q fragments: 4 chunks of K-dim 32 covering D=128
    Frag qa[4];
    #pragma unroll
    for (int c = 0; c < 4; ++c) {
        const __bf16* p = Qb + (size_t)(q0 + r) * HD + c * 32;
        qa[c].q[0] = *(const uint4*)(p + g * 8);
        qa[c].q[1] = *(const uint4*)(p + 16 + g * 8);
    }

    v8f acc[8];
    #pragma unroll
    for (int nt = 0; nt < 8; ++nt) acc[nt] = (v8f)0.0f;
    float rmax[8], rsum[8];
    #pragma unroll
    for (int v = 0; v < 8; ++v) { rmax[v] = -1.0e30f; rsum[v] = 0.0f; }

    const float scale = 0.08838834764831845f;   // 1/sqrt(128)

    for (int j0 = 0; j0 < SEQ; j0 += 32) {
        // ---- scores: two 16x16 tiles over this 32-key block
        v8f sc[2];
        sc[0] = (v8f)0.0f; sc[1] = (v8f)0.0f;
        #pragma unroll
        for (int t = 0; t < 2; ++t) {
            #pragma unroll
            for (int c = 0; c < 4; ++c) {
                Frag kb;
                const __bf16* p = Kb + (size_t)(j0 + t * 16 + r) * HD + c * 32;
                kb.q[0] = *(const uint4*)(p + g * 8);
                kb.q[1] = *(const uint4*)(p + 16 + g * 8);
                sc[t] = wmma_bf16(qa[c].v, kb.v, sc[t]);
            }
        }

        // ---- online softmax update + stage P tile (16x32) to LDS as bf16
        #pragma unroll
        for (int v = 0; v < 8; ++v) {
            float s0 = sc[0][v] * scale;
            float s1 = sc[1][v] * scale;
            float mt   = rowmax16(fmaxf(s0, s1));
            float mnew = fmaxf(rmax[v], mt);
            float alpha = __expf(rmax[v] - mnew);
            rmax[v] = mnew;
            float p0 = __expf(s0 - mnew);
            float p1 = __expf(s1 - mnew);
            rsum[v] = rsum[v] * alpha + rowsum16(p0 + p1);
            #pragma unroll
            for (int nt = 0; nt < 8; ++nt) acc[nt][v] *= alpha;
            int m = v + 8 * g;
            pw[m * 32 + r]      = (__bf16)p0;
            pw[m * 32 + 16 + r] = (__bf16)p1;
        }
        asm volatile("s_wait_dscnt 0" ::: "memory");   // wave-private LDS RAW

        // re-load P in A-fragment layout (16x32)
        Frag pa;
        pa.q[0] = *(const uint4*)(pw + r * 32 + g * 8);
        pa.q[1] = *(const uint4*)(pw + r * 32 + 16 + g * 8);

        // ---- PV: 8 d-tiles of 16, contraction over the 32 keys
        #pragma unroll
        for (int nt = 0; nt < 8; ++nt) {
            Frag vb;
            const __bf16* p = Vb + (size_t)(nt * 16 + r) * SEQ + j0;
            vb.q[0] = *(const uint4*)(p + g * 8);
            vb.q[1] = *(const uint4*)(p + 16 + g * 8);
            acc[nt] = wmma_bf16(pa.v, vb.v, acc[nt]);
        }
    }

    // ---- normalize + write token-major bf16 [B,S,E]
    #pragma unroll
    for (int nt = 0; nt < 8; ++nt)
        #pragma unroll
        for (int v = 0; v < 8; ++v) {
            int s = q0 + v + 8 * g;
            int e = h * HD + nt * 16 + r;
            O[((size_t)b * SEQ + s) * EMB + e] = (__bf16)(acc[nt][v] / rsum[v]);
        }
}

// ---------------------------------------------------------------------------
// Host-side launch
// ---------------------------------------------------------------------------
extern "C" void kernel_launch(void* const* d_in, const int* in_sizes, int n_in,
                              void* d_out, int out_size, void* d_ws, size_t ws_size,
                              hipStream_t stream) {
    const float* x  = (const float*)d_in[0];
    const float* Wq = (const float*)d_in[1];
    const float* Wk = (const float*)d_in[2];
    const float* Wv = (const float*)d_in[3];
    const float* Wo = (const float*)d_in[4];

    char* ws = (char*)d_ws;
    constexpr size_t SZ_XBF  = (size_t)MTOK * EMB * 2;      // 16.8 MB
    constexpr size_t SZ_W    = (size_t)EMB * EMB * 2;       //  8.4 MB each
    constexpr size_t SZ_TMPF = (size_t)MTOK * EMB * 4;      // 33.6 MB
    constexpr size_t SZ_HBF  = (size_t)MTOK * EMB * 2;      // 16.8 MB

    __bf16* xbf   = (__bf16*)(ws);
    __bf16* wqt   = (__bf16*)(ws + SZ_XBF);
    __bf16* wkt   = (__bf16*)(ws + SZ_XBF + SZ_W);
    __bf16* wvt   = (__bf16*)(ws + SZ_XBF + 2 * SZ_W);
    __bf16* wot   = (__bf16*)(ws + SZ_XBF + 3 * SZ_W);
    float*  tmpf  = (float*) (ws + SZ_XBF + 4 * SZ_W);
    __bf16* qbf   = (__bf16*)(ws + SZ_XBF + 4 * SZ_W + SZ_TMPF);
    __bf16* kbf   = (__bf16*)(ws + SZ_XBF + 4 * SZ_W + SZ_TMPF + SZ_HBF);
    __bf16* vtbf  = (__bf16*)(ws + SZ_XBF + 4 * SZ_W + SZ_TMPF + 2 * SZ_HBF);
    __bf16* atbf  = (__bf16*)(ws + SZ_XBF + 4 * SZ_W + SZ_TMPF + 3 * SZ_HBF);

    const int NX = MTOK * EMB;          // 8.4M
    const int NW = EMB * EMB;           // 4.2M

    cvt_f32_bf16<<<(NX + 255) / 256, 256, 0, stream>>>(x, xbf, NX);
    cvt_w_transpose<<<(NW + 255) / 256, 256, 0, stream>>>(Wq, wqt, EMB, EMB);
    cvt_w_transpose<<<(NW + 255) / 256, 256, 0, stream>>>(Wk, wkt, EMB, EMB);
    cvt_w_transpose<<<(NW + 255) / 256, 256, 0, stream>>>(Wv, wvt, EMB, EMB);
    cvt_w_transpose<<<(NW + 255) / 256, 256, 0, stream>>>(Wo, wot, EMB, EMB);

    dim3 ggrid(EMB / 128, MTOK / 64);   // (16, 64)
    const int NROPE = BATCH * NH * SEQ * (HD / 2);
    const int NVT   = BATCH * NH * HD * SEQ;

    // Q = rope(heads(x @ Wq))
    gemm_bf16<<<ggrid, 128, 0, stream>>>(xbf, wqt, tmpf, MTOK, EMB, EMB);
    rope_to_bf16<<<(NROPE + 255) / 256, 256, 0, stream>>>(tmpf, qbf);
    // K = rope(heads(x @ Wk))
    gemm_bf16<<<ggrid, 128, 0, stream>>>(xbf, wkt, tmpf, MTOK, EMB, EMB);
    rope_to_bf16<<<(NROPE + 255) / 256, 256, 0, stream>>>(tmpf, kbf);
    // V = heads(x @ Wv), stored transposed [B,H,D,S]
    gemm_bf16<<<ggrid, 128, 0, stream>>>(xbf, wvt, tmpf, MTOK, EMB, EMB);
    v_transpose<<<(NVT + 255) / 256, 256, 0, stream>>>(tmpf, vtbf);

    // flash attention -> bf16 [B,S,E]
    flash_attn<<<dim3(SEQ / 64, BATCH * NH), 128, 0, stream>>>(qbf, kbf, vtbf, atbf);

    // out = attn @ Wo  (fp32 result straight into d_out)
    gemm_bf16<<<ggrid, 128, 0, stream>>>(atbf, wot, (float*)d_out, MTOK, EMB, EMB);
}
